// PoolingAttention_84945863180824
// MI455X (gfx1250) — matrix operand order repaired and checked by reading, not compiled, not run on hardware
//
#include <hip/hip_runtime.h>
#include <stdint.h>

// ---------------------------------------------------------------------------
// Problem constants
// ---------------------------------------------------------------------------
constexpr int B_   = 16;
constexpr int N_   = 577;          // tokens
constexpr int C_   = 768;          // channels
constexpr int H_   = 12;           // heads
constexpr int DH   = 64;           // head dim
constexpr int NP   = 640;          // tokens padded (multiple of 64)
constexpr int KEEP = 403;          // int(576*0.7)
constexpr int NKEEP= 404;          // keep_num + 1 (cls token)

using u16 = unsigned short;
using u32 = unsigned int;
using u64 = unsigned long long;

typedef __attribute__((ext_vector_type(16))) __bf16 v16bf;
typedef __attribute__((ext_vector_type(8)))  float  v8f;
typedef __attribute__((ext_vector_type(4)))  u32    v4u;
typedef __attribute__((ext_vector_type(8)))  int    v8i;
typedef __attribute__((ext_vector_type(4)))  int    v4i;

// LDS layout for k_attn (bytes)
constexpr u32 S_OFF   = 0;                          // col-major scores 640x32 f32
constexpr u32 SB_OFF  = 32 * NP * 4;                // probs row-major 32x640 bf16
constexpr u32 RED_OFF = SB_OFF + 32 * NP * 2;       // 128 f32 reduction scratch
constexpr u32 KST_OFF = RED_OFF + 512;              // K stage 640x64 bf16 (TDM dest)
constexpr u32 SMEM_SZ = KST_OFF + NP * DH * 2;      // 205312 B  (< 320 KB)

// ---------------------------------------------------------------------------
// Helpers
// ---------------------------------------------------------------------------
__device__ inline u16 f2bf(float f) {                 // fp32 -> bf16 RNE
    u32 u = __builtin_bit_cast(u32, f);
    u32 r = u + 0x7FFFu + ((u >> 16) & 1u);
    return (u16)(r >> 16);
}
__device__ inline u32 pack2bf(float a, float b) {
    return (u32)f2bf(a) | ((u32)f2bf(b) << 16);
}

// A-fragment: 16x32 bf16, row-major source, row stride `ld` (elements).
// ISA layout: lane l (m=l&15, h=l>>4) holds K = {h*8..h*8+7} and {16+h*8..+7}.
__device__ inline v16bf load_a16x32(const u16* t, int ld) {
    int l = threadIdx.x & 31;
    const u16* p = t + (l & 15) * ld + (l >> 4) * 8;
    union { uint4 q[2]; v16bf v; } u;
    u.q[0] = *reinterpret_cast<const uint4*>(p);       // K group 0 (16B)
    u.q[1] = *reinterpret_cast<const uint4*>(p + 16);  // K group 1 (16B)
    return u.v;
}

// B-fragment: 32x16 bf16 loaded from B^T stored row-major N x K (stride ld).
// ISA layout: lane l (n=l&15, h=l>>4) holds K = {h*8..h*8+7} and {16+h*8..+7}.
__device__ inline v16bf load_b32x16T(const u16* t, int ld) {
    int l = threadIdx.x & 31;
    const u16* p = t + (l & 15) * ld + (l >> 4) * 8;
    union { uint4 q[2]; v16bf v; } u;
    u.q[0] = *reinterpret_cast<const uint4*>(p);
    u.q[1] = *reinterpret_cast<const uint4*>(p + 16);
    return u.v;
}

__device__ inline v8f wmma_bf16(v16bf a, v16bf b, v8f c) {
    return __builtin_amdgcn_wmma_f32_16x16x32_bf16(
        false, a, false, b, (short)0, c, false, false);
}

// ---------------------------------------------------------------------------
// K0a: x (B,577,768) fp32 -> xb (B,640,768) bf16, zero padded rows
// ---------------------------------------------------------------------------
__global__ void k_conv_x(const float* __restrict__ x, u16* __restrict__ xb) {
    int e = blockIdx.x * blockDim.x + threadIdx.x;     // exact multiple
    int c = e % C_;
    int r = (e / C_) % NP;
    int b = e / (C_ * NP);
    float v = (r < N_) ? x[((size_t)b * N_ + r) * C_ + c] : 0.0f;
    xb[e] = f2bf(v);
}

// K0b: transpose+convert W (K x N) fp32 -> W^T (N x K) bf16
__global__ void k_transpose(const float* __restrict__ src, u16* __restrict__ dst,
                            int K, int N) {
    int e = blockIdx.x * blockDim.x + threadIdx.x;
    if (e >= K * N) return;
    int k = e % K;
    int n = e / K;
    dst[e] = f2bf(src[(size_t)k * N + n]);             // dst[n*K+k]
}

__global__ void k_zero_f32(float* __restrict__ p, int n) {
    int e = blockIdx.x * blockDim.x + threadIdx.x;
    if (e < n) p[e] = 0.0f;
}

// ---------------------------------------------------------------------------
// K1: QKV GEMM.  One wave per block computes a 32(M) x 64(N) tile with a
// software-pipelined (ping-pong) K-loop.
// ---------------------------------------------------------------------------
__global__ void __launch_bounds__(32)
k_qkv(const u16* __restrict__ xb, const u16* __restrict__ wT,
      u16* __restrict__ qb, u16* __restrict__ kb, u16* __restrict__ vT) {
    int l  = threadIdx.x;
    int mt = blockIdx.x;                                // 0..19 (32-row tiles)
    int nt = blockIdx.y;                                // 0..35
    int b  = blockIdx.z;
    const u16* A  = xb + (size_t)b * NP * C_ + (size_t)mt * 32 * C_;
    const u16* Bt = wT + (size_t)nt * 64 * C_;

    v8f acc[8];
    #pragma unroll
    for (int i = 0; i < 8; ++i) acc[i] = (v8f){0,0,0,0,0,0,0,0};

    v16bf aP[2], bP[4], aQ[2], bQ[4];
    #pragma unroll
    for (int r = 0; r < 2; ++r) aP[r] = load_a16x32(A + r * 16 * C_, C_);
    #pragma unroll
    for (int i = 0; i < 4; ++i) bP[i] = load_b32x16T(Bt + i * 16 * C_, C_);

    for (int kk = 0; kk < C_; kk += 64) {
        int k1 = kk + 32;                                    // always < C_
        #pragma unroll
        for (int r = 0; r < 2; ++r) aQ[r] = load_a16x32(A + r * 16 * C_ + k1, C_);
        #pragma unroll
        for (int i = 0; i < 4; ++i) bQ[i] = load_b32x16T(Bt + i * 16 * C_ + k1, C_);
        #pragma unroll
        for (int r = 0; r < 2; ++r)
            #pragma unroll
            for (int i = 0; i < 4; ++i)
                acc[r * 4 + i] = wmma_bf16(aP[r], bP[i], acc[r * 4 + i]);

        int k2 = (kk + 64 < C_) ? kk + 64 : 0;               // dummy on last
        #pragma unroll
        for (int r = 0; r < 2; ++r) aP[r] = load_a16x32(A + r * 16 * C_ + k2, C_);
        #pragma unroll
        for (int i = 0; i < 4; ++i) bP[i] = load_b32x16T(Bt + i * 16 * C_ + k2, C_);
        #pragma unroll
        for (int r = 0; r < 2; ++r)
            #pragma unroll
            for (int i = 0; i < 4; ++i)
                acc[r * 4 + i] = wmma_bf16(aQ[r], bQ[i], acc[r * 4 + i]);
    }

    int gcol0 = nt * 64;
    int s = gcol0 / C_;                                 // 0=q 1=k 2=v
    int h = (gcol0 % C_) / DH;
    size_t bh = (size_t)(b * H_ + h);
    int colL = l & 15, halfM = l >> 4;

    if (s == 2) {
        // v stored transposed: lane's 8 values are consecutive n -> b128 store
        #pragma unroll
        for (int rs = 0; rs < 2; ++rs)
            #pragma unroll
            for (int i = 0; i < 4; ++i) {
                v8f a = acc[rs * 4 + i];
                int d  = i * 16 + colL;
                int nb = mt * 32 + rs * 16 + 8 * halfM;
                uint4 pk = { pack2bf(a[0], a[1]), pack2bf(a[2], a[3]),
                             pack2bf(a[4], a[5]), pack2bf(a[6], a[7]) };
                *reinterpret_cast<uint4*>(vT + (bh * DH + d) * NP + nb) = pk;
            }
    } else {
        u16* dst  = (s == 0) ? qb : kb;
        float sc  = (s == 0) ? 0.125f : 1.0f;           // fold dh^-0.5 into q
        #pragma unroll
        for (int rs = 0; rs < 2; ++rs)
            #pragma unroll
            for (int i = 0; i < 4; ++i)
                #pragma unroll
                for (int r = 0; r < 8; ++r) {
                    int n = mt * 32 + rs * 16 + r + 8 * halfM;
                    int d = i * 16 + colL;
                    dst[(bh * NP + n) * DH + d] = f2bf(acc[rs * 4 + i][r] * sc);
                }
    }
}

// ---------------------------------------------------------------------------
// K2: fused attention.  block = 128 thr (4 waves) owns 32 query rows x (b,h).
// The full K matrix for this (b,h) (640x64 bf16 = 80KB) is DMA'd into LDS by
// the Tensor Data Mover; scores live col-major in LDS; probs bf16 in LDS.
// ---------------------------------------------------------------------------
__global__ void __launch_bounds__(128)
k_attn(const u16* __restrict__ qb, const u16* __restrict__ kb,
       const u16* __restrict__ vT, u16* __restrict__ ao,
       float* __restrict__ aw) {
    extern __shared__ char smem[];
    float* S   = (float*)(smem + S_OFF);                 // col-major 640x32
    u16*   Sb  = (u16*)(smem + SB_OFF);                  // row-major 32x640
    float* red = (float*)(smem + RED_OFF);               // 128 f32
    u16*   Kl  = (u16*)(smem + KST_OFF);                 // staged K 640x64

    int tid = threadIdx.x, w = tid >> 5, l = tid & 31;
    int rb = blockIdx.x;                                 // 0..19 row blocks
    int h  = blockIdx.y;
    int b  = blockIdx.z;
    size_t bh = (size_t)(b * H_ + h);
    const u16* Q = qb + (bh * NP + (size_t)rb * 32) * DH;
    const u16* K = kb + bh * NP * DH;
    const u16* V = vT + bh * DH * NP;

    int colL = l & 15, halfM = l >> 4;

    // ---- TDM: stage K (640x64 bf16) into LDS, issued once by wave 0 ----
#if __has_builtin(__builtin_amdgcn_tensor_load_to_lds)
    if (w == 0) {
        u64 ga = (u64)(uintptr_t)K;
        v4u g0 = { 0x1u,                                   // count=1, user mode
                   (u32)KST_OFF,                           // lds_addr (bytes)
                   (u32)ga,                                 // global_addr[31:0]
                   ((u32)(ga >> 32) & 0x01FFFFFFu) | (2u << 30) }; // [56:32]|type=2
        v8i g1 = { (int)0x00010000,      // data_size=1 (2B), no mask/pad/iter
                   (int)(64u  << 16),    // tensor_dim0      = 64   [79:48]
                   (int)(640u << 16),    // tensor_dim1      = 640  [111:80]
                   (int)(64u  << 16),    // tile_dim0        = 64   [127:112]
                   (int)640,             // tile_dim1        = 640  [143:128]
                   (int)64,              // tensor_dim0_stride = 64 [207:160]
                   0, 0 };
        v4i gz4 = {0, 0, 0, 0};          // groups 2/3 unused (2-D tile)
        v8i gz8 = {0, 0, 0, 0, 0, 0, 0, 0};
        __builtin_amdgcn_tensor_load_to_lds(g0, g1, gz4, gz4, gz8, 0);
    }
#else
    {   // fallback: cooperative copy K -> LDS
        const uint4* src = (const uint4*)K;
        uint4* dst = (uint4*)Kl;
        for (int i = tid; i < NP * DH * 2 / 16; i += 128) dst[i] = src[i];
    }
#endif

    // ---- overlap: q fragments while the DMA runs ----
    v16bf qf[2][2];
    #pragma unroll
    for (int rs = 0; rs < 2; ++rs)
        #pragma unroll
        for (int kh = 0; kh < 2; ++kh)
            qf[rs][kh] = load_a16x32(Q + rs * 16 * DH + kh * 32, DH);

#if __has_builtin(__builtin_amdgcn_tensor_load_to_lds)
    if (w == 0) __builtin_amdgcn_s_wait_tensorcnt(0);
#endif
    __syncthreads();                                     // K staged for all waves

    // ---- phase A: S = q_block @ k^T, B-fragments from LDS ----
    for (int ct = w; ct < NP / 16; ct += 4) {
        v16bf kf0 = load_b32x16T(Kl + ct * 16 * DH + 0,  DH);   // ds_read_b128
        v16bf kf1 = load_b32x16T(Kl + ct * 16 * DH + 32, DH);
        int col = ct * 16 + colL;
        bool ok = col < N_;
        #pragma unroll
        for (int rs = 0; rs < 2; ++rs) {
            v8f c = {0,0,0,0,0,0,0,0};
            c = wmma_bf16(qf[rs][0], kf0, c);
            c = wmma_bf16(qf[rs][1], kf1, c);
            float v0 = ok ? c[0] : -1.0e30f, v1 = ok ? c[1] : -1.0e30f;
            float v2 = ok ? c[2] : -1.0e30f, v3 = ok ? c[3] : -1.0e30f;
            float v4 = ok ? c[4] : -1.0e30f, v5 = ok ? c[5] : -1.0e30f;
            float v6 = ok ? c[6] : -1.0e30f, v7 = ok ? c[7] : -1.0e30f;
            // col-major: lane's 8 rows are contiguous -> two b128 LDS stores
            int idx = col * 32 + rs * 16 + 8 * halfM;
            *reinterpret_cast<float4*>(S + idx)     = (float4){v0, v1, v2, v3};
            *reinterpret_cast<float4*>(S + idx + 4) = (float4){v4, v5, v6, v7};
            // token scoring: sum_h |attn[b,h,0,col]| (pre-softmax, scaled)
            if (rb == 0 && rs == 0 && halfM == 0 && ok)
                atomicAdd(&aw[b * NP + col], fabsf(v0));
        }
    }
    __syncthreads();

    // ---- phase B: softmax rows (4 threads per row; S is col-major) ----
    {
        int row = tid >> 2, j = tid & 3;
        float m = -3.0e38f;
        for (int c = j; c < NP; c += 4) m = fmaxf(m, S[c * 32 + row]);
        red[tid] = m; __syncthreads();
        float m4 = fmaxf(fmaxf(red[row * 4 + 0], red[row * 4 + 1]),
                         fmaxf(red[row * 4 + 2], red[row * 4 + 3]));
        __syncthreads();
        float sum = 0.0f;
        for (int c = j; c < NP; c += 4) {
            float e = __expf(S[c * 32 + row] - m4);
            S[c * 32 + row] = e; sum += e;
        }
        red[tid] = sum; __syncthreads();
        float inv = 1.0f / (red[row * 4 + 0] + red[row * 4 + 1] +
                            red[row * 4 + 2] + red[row * 4 + 3]);
        u16* Sbr = Sb + row * NP;
        for (int c = j; c < NP; c += 4) Sbr[c] = f2bf(S[c * 32 + row] * inv);
    }
    __syncthreads();

    // ---- phase C: O = prob(32x640) @ v(640x64).  Wave w owns N-tile w for
    // BOTH row sub-tiles (shared V fragments), double-buffered V loads. ----
    {
        int nt = w;
        v8f acc0 = {0,0,0,0,0,0,0,0}, acc1 = acc0;
        v16bf bfP = load_b32x16T(V + nt * 16 * NP, NP);
        for (int kk = 0; kk < NP; kk += 32) {
            int kn = (kk + 32 < NP) ? kk + 32 : 0;        // dummy on last
            v16bf bfQ = load_b32x16T(V + nt * 16 * NP + kn, NP);
            v16bf af0 = load_a16x32(Sb + kk, NP);                // ds_read
            v16bf af1 = load_a16x32(Sb + 16 * NP + kk, NP);      // ds_read
            acc0 = wmma_bf16(af0, bfP, acc0);
            acc1 = wmma_bf16(af1, bfP, acc1);
            bfP = bfQ;
        }
        #pragma unroll
        for (int r = 0; r < 8; ++r) {
            int d  = nt * 16 + colL;
            int n0 = rb * 32 + 0  + r + 8 * halfM;
            int n1 = rb * 32 + 16 + r + 8 * halfM;
            ao[((size_t)b * NP + n0) * C_ + h * DH + d] = f2bf(acc0[r]);
            ao[((size_t)b * NP + n1) * C_ + h * DH + d] = f2bf(acc1[r]);
        }
    }
}

// ---------------------------------------------------------------------------
// K3: projection GEMM + bias -> fp32 output (rows >= 577 dropped).
// ---------------------------------------------------------------------------
__global__ void __launch_bounds__(32)
k_proj(const u16* __restrict__ ao, const u16* __restrict__ wT,
       const float* __restrict__ bias, float* __restrict__ out) {
    int l  = threadIdx.x;
    int mt = blockIdx.x;                                // 0..19
    int nt = blockIdx.y;                                // 0..11
    int b  = blockIdx.z;
    const u16* A  = ao + ((size_t)b * NP + (size_t)mt * 32) * C_;
    const u16* Bt = wT + (size_t)nt * 64 * C_;

    v8f acc[8];
    #pragma unroll
    for (int i = 0; i < 8; ++i) acc[i] = (v8f){0,0,0,0,0,0,0,0};

    v16bf aP[2], bP[4], aQ[2], bQ[4];
    #pragma unroll
    for (int r = 0; r < 2; ++r) aP[r] = load_a16x32(A + r * 16 * C_, C_);
    #pragma unroll
    for (int i = 0; i < 4; ++i) bP[i] = load_b32x16T(Bt + i * 16 * C_, C_);

    for (int kk = 0; kk < C_; kk += 64) {
        int k1 = kk + 32;
        #pragma unroll
        for (int r = 0; r < 2; ++r) aQ[r] = load_a16x32(A + r * 16 * C_ + k1, C_);
        #pragma unroll
        for (int i = 0; i < 4; ++i) bQ[i] = load_b32x16T(Bt + i * 16 * C_ + k1, C_);
        #pragma unroll
        for (int r = 0; r < 2; ++r)
            #pragma unroll
            for (int i = 0; i < 4; ++i)
                acc[r * 4 + i] = wmma_bf16(aP[r], bP[i], acc[r * 4 + i]);

        int k2 = (kk + 64 < C_) ? kk + 64 : 0;
        #pragma unroll
        for (int r = 0; r < 2; ++r) aP[r] = load_a16x32(A + r * 16 * C_ + k2, C_);
        #pragma unroll
        for (int i = 0; i < 4; ++i) bP[i] = load_b32x16T(Bt + i * 16 * C_ + k2, C_);
        #pragma unroll
        for (int r = 0; r < 2; ++r)
            #pragma unroll
            for (int i = 0; i < 4; ++i)
                acc[r * 4 + i] = wmma_bf16(aQ[r], bQ[i], acc[r * 4 + i]);
    }

    int colL = l & 15, halfM = l >> 4;
    #pragma unroll
    for (int rs = 0; rs < 2; ++rs)
        #pragma unroll
        for (int i = 0; i < 4; ++i)
            #pragma unroll
            for (int r = 0; r < 8; ++r) {
                int n = mt * 32 + rs * 16 + r + 8 * halfM;
                if (n < N_) {
                    int col = nt * 64 + i * 16 + colL;
                    out[((size_t)b * N_ + n) * C_ + col] =
                        acc[rs * 4 + i][r] + bias[col];
                }
            }
}

// ---------------------------------------------------------------------------
// K4: top-k(403) of w = aw[b,1:577] by rank counting; sorted indices + cls 0.
// ---------------------------------------------------------------------------
__global__ void k_topk(const float* __restrict__ aw, int* __restrict__ idxb) {
    __shared__ float sw[576];
    __shared__ unsigned char sk[576];
    int b = blockIdx.x, t = threadIdx.x;                 // 576 threads
    sw[t] = aw[b * NP + 1 + t];
    __syncthreads();
    float v = sw[t];
    int rank = 0;
    for (int i = 0; i < 576; ++i) {
        float vi = sw[i];
        rank += (vi > v) || (vi == v && i < t);
    }
    int keep = rank < KEEP;
    sk[t] = (unsigned char)keep;
    __syncthreads();
    int pos = 0;
    for (int i = 0; i < t; ++i) pos += sk[i];
    if (keep) idxb[b * NKEEP + 1 + pos] = t + 1;
    if (t == 0) idxb[b * NKEEP] = 0;
}

// K5: keep_index[b, r, :] = idx[b, r]  (int32 written into d_out tail)
__global__ void k_bcast(const int* __restrict__ idxb, int* __restrict__ out) {
    int e = blockIdx.x * blockDim.x + threadIdx.x;
    if (e < B_ * NKEEP * C_) out[e] = idxb[e / C_];
}

// ---------------------------------------------------------------------------
// Launch
// ---------------------------------------------------------------------------
extern "C" void kernel_launch(void* const* d_in, const int* in_sizes, int n_in,
                              void* d_out, int out_size, void* d_ws, size_t ws_size,
                              hipStream_t stream) {
    const float* x      = (const float*)d_in[0];
    const float* W_qkv  = (const float*)d_in[1];
    const float* W_proj = (const float*)d_in[2];
    const float* b_proj = (const float*)d_in[3];

    char* ws = (char*)d_ws;
    auto take = [&](size_t bytes) {
        char* p = ws; ws += (bytes + 255) & ~size_t(255); return p;
    };
    u16*   xb     = (u16*)  take((size_t)B_ * NP * C_ * 2);
    u16*   WqkvT  = (u16*)  take((size_t)3 * C_ * C_ * 2);
    u16*   WprojT = (u16*)  take((size_t)C_ * C_ * 2);
    u16*   qb     = (u16*)  take((size_t)B_ * H_ * NP * DH * 2);
    u16*   kb     = (u16*)  take((size_t)B_ * H_ * NP * DH * 2);
    u16*   vT     = (u16*)  take((size_t)B_ * H_ * NP * DH * 2);
    u16*   ao     = (u16*)  take((size_t)B_ * NP * C_ * 2);
    float* aw     = (float*)take((size_t)B_ * NP * 4);
    int*   idxb   = (int*)  take((size_t)B_ * NKEEP * 4);

    float* out0   = (float*)d_out;                                 // (16,577,768) f32
    int*   keepO  = (int*)d_out + (size_t)B_ * N_ * C_;            // (16,404,768) i32

    // prep
    k_conv_x   <<<(B_ * NP * C_) / 256, 256, 0, stream>>>(x, xb);
    k_transpose<<<(C_ * 3 * C_) / 256, 256, 0, stream>>>(W_qkv,  WqkvT,  C_, 3 * C_);
    k_transpose<<<(C_ * C_)     / 256, 256, 0, stream>>>(W_proj, WprojT, C_, C_);
    k_zero_f32 <<<(B_ * NP + 255) / 256, 256, 0, stream>>>(aw, B_ * NP);

    // qkv gemm (32x64 tiles per wave)
    k_qkv<<<dim3(NP / 32, 3 * C_ / 64, B_), 32, 0, stream>>>(xb, WqkvT, qb, kb, vT);

    // fused attention (TDM-staged K, LDS-resident scores)
    k_attn<<<dim3(NP / 32, H_, B_), 128, SMEM_SZ, stream>>>(qb, kb, vT, ao, aw);

    // projection
    k_proj<<<dim3(NP / 32, C_ / 64, B_), 32, 0, stream>>>(ao, WprojT, b_proj, out0);

    // token selection
    k_topk <<<B_, 576, 0, stream>>>(aw, idxb);
    k_bcast<<<(B_ * NKEEP * C_ + 255) / 256, 256, 0, stream>>>(idxb, keepO);
}
